// CrossAttention_34900904248065
// MI455X (gfx1250) — matrix operand order repaired and checked by reading, not compiled, not run on hardware
//
#include <hip/hip_runtime.h>

// ---------------------------------------------------------------------------
// CrossAttention for MI455X (gfx1250): bf16 WMMA everywhere.
//   q  = x_q @ W_q          -> scattered to [B,H,Tq,D]   (bf16)
//   kv = x_kv @ W_kv        -> k [B,H,Tkv,D], v^T [B,H,D,Tkv] (bf16)
//   y  = softmax((qk + bias)/SCALE) @ v   (flash-style, per-wave 16 q rows)
//   out = y @ W_proj        -> fp32
// ---------------------------------------------------------------------------

typedef __attribute__((ext_vector_type(16))) __bf16 v16bf;
typedef __attribute__((ext_vector_type(8)))  float  v8f;
typedef __attribute__((ext_vector_type(4)))  __bf16 v4bf;

union Frag16 { int4 q[2]; v16bf v; };   // 32 bytes: one WMMA bf16 A/B fragment per lane

constexpr int BATCH = 2;
constexpr int TQ    = 2048;
constexpr int TKV   = 2048;
constexpr int HEADS = 16;
constexpr int CDIM  = 1024;
constexpr int DHEAD = 64;
// reference: attn = (qk + bias) / SCALE, SCALE = DHEAD^-0.5 -> multiply by 8
constexpr float INV_SCALE = 8.0f;

__device__ __forceinline__ __bf16 f2bf(float f) { return (__bf16)f; }

__device__ __forceinline__ v8f wmma_bf16(v16bf a, v16bf b, v8f c) {
  // (neg_a, A, neg_b, B, c_mod, C, reuse_a, reuse_b)
  return __builtin_amdgcn_wmma_f32_16x16x32_bf16(false, a, false, b, (short)0, c,
                                                 false, false);
}

// ---------------------------------------------------------------------------
// DPP-based reductions across each 16-lane row (VALU pipe, no LDS round-trip).
// xor1 = quad_perm[1,0,3,2]=0xB1, xor2 = quad_perm[2,3,0,1]=0x4E,
// xor4 = DPP8 sel(i^4), xor8 = row_mirror(0x140) — valid because after the
// first three steps every 8-lane group is uniform and 15-L is in the other
// 8-group of the same 16-lane row.
// ---------------------------------------------------------------------------
template <int CTRL>
__device__ __forceinline__ float dpp_bcast(float x) {
  int v = __builtin_amdgcn_update_dpp(0, __float_as_int(x), CTRL, 0xf, 0xf, true);
  return __int_as_float(v);
}
__device__ __forceinline__ float dpp8_xor4(float x) {
  constexpr int sel = (4) | (5 << 3) | (6 << 6) | (7 << 9) |
                      (0 << 12) | (1 << 15) | (2 << 18) | (3 << 21);
  return __int_as_float(__builtin_amdgcn_mov_dpp8(__float_as_int(x), sel));
}
__device__ __forceinline__ float red_max16(float x) {
  x = fmaxf(x, dpp_bcast<0xB1>(x));
  x = fmaxf(x, dpp_bcast<0x4E>(x));
  x = fmaxf(x, dpp8_xor4(x));
  x = fmaxf(x, dpp_bcast<0x140>(x));
  return x;
}
__device__ __forceinline__ float red_sum16(float x) {
  x += dpp_bcast<0xB1>(x);
  x += dpp_bcast<0x4E>(x);
  x += dpp8_xor4(x);
  x += dpp_bcast<0x140>(x);
  return x;
}

// ---------------------------------------------------------------------------
// fp32 -> bf16 conversion, 4 elements per thread iteration
// ---------------------------------------------------------------------------
__global__ void cvt_f32_bf16_kernel(const float* __restrict__ in,
                                    __bf16* __restrict__ out, int n4) {
  int i = blockIdx.x * blockDim.x + threadIdx.x;
  int stride = gridDim.x * blockDim.x;
  for (; i < n4; i += stride) {
    float4 f = ((const float4*)in)[i];
    v4bf r;
    r[0] = f2bf(f.x); r[1] = f2bf(f.y); r[2] = f2bf(f.z); r[3] = f2bf(f.w);
    ((v4bf*)out)[i] = r;
  }
}

// ---------------------------------------------------------------------------
// Tiled bf16 GEMM: C(MxN) = A(MxK,row) * B(KxN,row).
// Block tile 128x128, 8 waves, each wave 32x64 (2x4 WMMA tiles), K step 32.
// MODE 0: fp32 C out (NT store).  MODE 1: q scatter.  MODE 2: k + v^T scatter.
// ---------------------------------------------------------------------------
template <int MODE>
__global__ __launch_bounds__(256)
void gemm_bf16_kernel(const __bf16* __restrict__ A, const __bf16* __restrict__ Bm,
                      float* __restrict__ Cf,
                      __bf16* __restrict__ O1, __bf16* __restrict__ O2,
                      int M, int N, int K) {
  __shared__ __align__(16) __bf16 As[128 * 32];   // [row][k]
  __shared__ __align__(16) __bf16 Bs[128 * 32];   // transposed: [n][k]

  const int tid  = threadIdx.x;
  const int m0   = blockIdx.y * 128;
  const int n0   = blockIdx.x * 128;
  const int lane = tid & 31;
  const int wid  = tid >> 5;
  const int wm   = (wid >> 1) * 32;
  const int wn   = (wid & 1) * 64;
  const int half = lane >> 4;
  const int l16  = lane & 15;

  v8f acc[2][4] = {};

  const int arow = tid >> 1;
  const int akp  = (tid & 1) * 16;
  const int bkr  = tid >> 3;
  const int bnn  = (tid & 7) * 16;

  for (int k0 = 0; k0 < K; k0 += 32) {
    {  // A tile -> LDS row-major (each thread moves 32 contiguous bytes)
      const int4* src = (const int4*)(A + (size_t)(m0 + arow) * K + k0 + akp);
      int4 a0 = src[0], a1 = src[1];
      int4* dst = (int4*)(As + arow * 32 + akp);
      dst[0] = a0; dst[1] = a1;
    }
    {  // B tile -> LDS transposed [n][k] so B fragments are contiguous
      Frag16 f;
      const int4* src = (const int4*)(Bm + (size_t)(k0 + bkr) * N + n0 + bnn);
      f.q[0] = src[0]; f.q[1] = src[1];
#pragma unroll
      for (int e = 0; e < 16; ++e) Bs[(bnn + e) * 32 + bkr] = f.v[e];
    }
    __syncthreads();

    // A fragment: lanes 0-15 hold K 0..15, lanes 16-31 hold K 16..31
    Frag16 af[2], bfr[4];
#pragma unroll
    for (int mt = 0; mt < 2; ++mt) {
      const __bf16* p = As + (wm + mt * 16 + l16) * 32 + half * 16;
      af[mt].q[0] = *(const int4*)p;
      af[mt].q[1] = *(const int4*)(p + 8);
    }
    // B fragment: lanes 0-15 hold K 0..15, lanes 16-31 hold K 16..31 (contig in Bs)
#pragma unroll
    for (int nt = 0; nt < 4; ++nt) {
      const __bf16* p = Bs + (wn + nt * 16 + l16) * 32 + half * 16;
      bfr[nt].q[0] = *(const int4*)p;
      bfr[nt].q[1] = *(const int4*)(p + 8);
    }
#pragma unroll
    for (int mt = 0; mt < 2; ++mt)
#pragma unroll
      for (int nt = 0; nt < 4; ++nt)
        acc[mt][nt] = wmma_bf16(af[mt].v, bfr[nt].v, acc[mt][nt]);

    __syncthreads();
  }

  // Epilogue. C layout: element (vgpr v, lane) -> m = v + 8*(lane>>4), n = lane&15
#pragma unroll
  for (int mt = 0; mt < 2; ++mt) {
#pragma unroll
    for (int nt = 0; nt < 4; ++nt) {
#pragma unroll
      for (int v = 0; v < 8; ++v) {
        int row = m0 + wm + mt * 16 + v + half * 8;
        int col = n0 + wn + nt * 16 + l16;
        float val = acc[mt][nt][v];
        if constexpr (MODE == 0) {
          // final output: written once, never re-read -> non-temporal
          __builtin_nontemporal_store(val, Cf + (size_t)row * N + col);
        } else if constexpr (MODE == 1) {
          int b = row >> 11, t = row & (TQ - 1);
          int h = col >> 6, d = col & 63;
          O1[(((size_t)(b * HEADS + h)) * TQ + t) * DHEAD + d] = f2bf(val);
        } else {
          int b = row >> 11, t = row & (TKV - 1);
          if (col < CDIM) {  // k part -> [B,H,Tkv,D]
            int h = col >> 6, d = col & 63;
            O1[(((size_t)(b * HEADS + h)) * TKV + t) * DHEAD + d] = f2bf(val);
          } else {           // v part -> transposed [B,H,D,Tkv]
            int c = col - CDIM;
            int h = c >> 6, d = c & 63;
            O2[(((size_t)(b * HEADS + h)) * DHEAD + d) * TKV + t] = f2bf(val);
          }
        }
      }
    }
  }
}

// ---------------------------------------------------------------------------
// Flash attention: block = 4 waves x 16 q rows = 64 q rows, kv chunks of 64.
// K/V stay L2-resident (reused by all q tiles of a head); the 268MB bias
// stream is read once -> non-temporal loads so it doesn't evict K/V.
// ---------------------------------------------------------------------------
__global__ __launch_bounds__(128)
void attn_kernel(const __bf16* __restrict__ Q, const __bf16* __restrict__ Kh,
                 const __bf16* __restrict__ Vt, const float* __restrict__ bias,
                 __bf16* __restrict__ Y) {
  __shared__ __align__(16) __bf16 Pbuf[4][16 * 64];  // wave-private P tiles

  const int tid  = threadIdx.x;
  const int lane = tid & 31, wid = tid >> 5;
  const int half = lane >> 4, l16 = lane & 15;
  const int bz = blockIdx.z, h = blockIdx.y;
  const int q0 = blockIdx.x * 64 + wid * 16;
  const size_t bh = (size_t)(bz * HEADS + h);

  const __bf16* qh = Q + bh * TQ * DHEAD;
  const __bf16* kh = Kh + bh * TKV * DHEAD;
  const __bf16* vh = Vt + bh * DHEAD * TKV;
  const float*  bb = bias + ((size_t)h * TQ + q0) * TKV;  // bias batch dim == 1

  // Q A-fragments, D=64 split into two K=32 steps
  Frag16 qa[2];
#pragma unroll
  for (int d0 = 0; d0 < 2; ++d0) {
    const __bf16* p = qh + (size_t)(q0 + l16) * DHEAD + d0 * 32 + half * 16;
    qa[d0].q[0] = *(const int4*)p;
    qa[d0].q[1] = *(const int4*)(p + 8);
  }

  float m_run[8], l_run[8];
#pragma unroll
  for (int v = 0; v < 8; ++v) { m_run[v] = -1e30f; l_run[v] = 0.0f; }
  v8f o[4] = {};

  __bf16* pb = &Pbuf[wid][0];

  for (int j = 0; j < TKV / 64; ++j) {
    // ---- S = Q @ K^T for a 16x64 score tile ----
    v8f s[4] = {};
#pragma unroll
    for (int d0 = 0; d0 < 2; ++d0) {
#pragma unroll
      for (int nt = 0; nt < 4; ++nt) {
        Frag16 kf;  // B operand: column n = kv row, K-dim = d (contig in k_bf)
        const __bf16* p =
            kh + (size_t)(j * 64 + nt * 16 + l16) * DHEAD + d0 * 32 + half * 16;
        kf.q[0] = *(const int4*)p;
        kf.q[1] = *(const int4*)(p + 8);
        s[nt] = wmma_bf16(qa[d0].v, kf.v, s[nt]);
      }
    }
    // ---- bias, scale, online-softmax stats (rows live across 16-lane halves)
    float mnew[8];
#pragma unroll
    for (int v = 0; v < 8; ++v) {
      const int mloc = v + half * 8;
      float bm = -1e30f;
#pragma unroll
      for (int nt = 0; nt < 4; ++nt) {
        float bv = __builtin_nontemporal_load(
            bb + (size_t)mloc * TKV + j * 64 + nt * 16 + l16);
        float sv = (s[nt][v] + bv) * INV_SCALE;
        s[nt][v] = sv;
        bm = fmaxf(bm, sv);
      }
      bm = red_max16(bm);
      mnew[v] = fmaxf(m_run[v], bm);
    }
#pragma unroll
    for (int v = 0; v < 8; ++v) {
      float alpha = __expf(m_run[v] - mnew[v]);
      float rs = 0.0f;
#pragma unroll
      for (int nt = 0; nt < 4; ++nt) {
        float p = __expf(s[nt][v] - mnew[v]);
        s[nt][v] = p;
        rs += p;
      }
      rs = red_sum16(rs);
      l_run[v] = l_run[v] * alpha + rs;
      m_run[v] = mnew[v];
#pragma unroll
      for (int dt = 0; dt < 4; ++dt) o[dt][v] *= alpha;
    }
    // ---- P (C layout) -> LDS -> A layout ----
#pragma unroll
    for (int v = 0; v < 8; ++v) {
      const int mloc = v + half * 8;
#pragma unroll
      for (int nt = 0; nt < 4; ++nt)
        pb[mloc * 64 + nt * 16 + l16] = f2bf(s[nt][v]);
    }
    asm volatile("s_wait_dscnt 0" ::: "memory");  // wave-private LDS RAW fence
    // ---- O += P @ V  (K-dim = kv chunk, two steps of 32) ----
#pragma unroll
    for (int kk = 0; kk < 2; ++kk) {
      Frag16 pa;
      const __bf16* pr = pb + l16 * 64 + kk * 32 + half * 16;
      pa.q[0] = *(const int4*)pr;
      pa.q[1] = *(const int4*)(pr + 8);
#pragma unroll
      for (int dt = 0; dt < 4; ++dt) {
        Frag16 vf;  // B operand: column n = d, K-dim = kv (contig in v^T)
        const __bf16* vp =
            vh + (size_t)(dt * 16 + l16) * TKV + j * 64 + kk * 32 + half * 16;
        vf.q[0] = *(const int4*)vp;
        vf.q[1] = *(const int4*)(vp + 8);
        o[dt] = wmma_bf16(pa.v, vf.v, o[dt]);
      }
    }
  }

  // ---- normalize and scatter y -> [B,Tq,C] (bf16) ----
#pragma unroll
  for (int v = 0; v < 8; ++v) {
    const float inv = 1.0f / l_run[v];
    const int mloc = v + half * 8;
#pragma unroll
    for (int dt = 0; dt < 4; ++dt) {
      size_t idx = ((size_t)bz * TQ + q0 + mloc) * CDIM + h * DHEAD + dt * 16 + l16;
      Y[idx] = f2bf(o[dt][v] * inv);
    }
  }
}

// ---------------------------------------------------------------------------
extern "C" void kernel_launch(void* const* d_in, const int* in_sizes, int n_in,
                              void* d_out, int out_size, void* d_ws, size_t ws_size,
                              hipStream_t stream) {
  const float* x_q  = (const float*)d_in[0];
  const float* x_kv = (const float*)d_in[1];
  const float* bias = (const float*)d_in[2];
  const float* W_q  = (const float*)d_in[3];
  const float* W_kv = (const float*)d_in[4];
  const float* W_p  = (const float*)d_in[5];

  char* ws = (char*)d_ws;
  size_t off = 0;
  auto take = [&](size_t elems) -> void* {
    void* p = ws + off;
    off += (elems * sizeof(__bf16) + 255) & ~(size_t)255;
    return p;
  };

  const size_t nXQ  = (size_t)BATCH * TQ * CDIM;    // 4 Mi elems
  const size_t nXKV = (size_t)BATCH * TKV * CDIM;
  const size_t nWQ  = (size_t)CDIM * CDIM;
  const size_t nWKV = (size_t)CDIM * 2 * CDIM;
  const size_t nWP  = (size_t)CDIM * CDIM;

  __bf16* xq_bf  = (__bf16*)take(nXQ);
  __bf16* xkv_bf = (__bf16*)take(nXKV);
  __bf16* wq_bf  = (__bf16*)take(nWQ);
  __bf16* wkv_bf = (__bf16*)take(nWKV);
  __bf16* wp_bf  = (__bf16*)take(nWP);
  __bf16* q_bf   = (__bf16*)take(nXQ);    // [B,H,Tq,D]
  __bf16* k_bf   = (__bf16*)take(nXKV);   // [B,H,Tkv,D]
  __bf16* vt_bf  = (__bf16*)take(nXKV);   // [B,H,D,Tkv]
  __bf16* y_bf   = (__bf16*)take(nXQ);    // [B,Tq,C]

  cvt_f32_bf16_kernel<<<2048, 256, 0, stream>>>(x_q,  xq_bf,  (int)(nXQ / 4));
  cvt_f32_bf16_kernel<<<2048, 256, 0, stream>>>(x_kv, xkv_bf, (int)(nXKV / 4));
  cvt_f32_bf16_kernel<<<512,  256, 0, stream>>>(W_q,  wq_bf,  (int)(nWQ / 4));
  cvt_f32_bf16_kernel<<<1024, 256, 0, stream>>>(W_kv, wkv_bf, (int)(nWKV / 4));
  cvt_f32_bf16_kernel<<<512,  256, 0, stream>>>(W_p,  wp_bf,  (int)(nWP / 4));

  // q = x_q @ W_q, scatter to heads
  gemm_bf16_kernel<1><<<dim3(CDIM / 128, (BATCH * TQ) / 128), 256, 0, stream>>>(
      xq_bf, wq_bf, nullptr, q_bf, nullptr, BATCH * TQ, CDIM, CDIM);
  // kv = x_kv @ W_kv, scatter k and transposed v
  gemm_bf16_kernel<2><<<dim3(2 * CDIM / 128, (BATCH * TKV) / 128), 256, 0, stream>>>(
      xkv_bf, wkv_bf, nullptr, k_bf, vt_bf, BATCH * TKV, 2 * CDIM, CDIM);
  // flash attention
  attn_kernel<<<dim3(TQ / 64, HEADS, BATCH), 128, 0, stream>>>(
      q_bf, k_bf, vt_bf, bias, y_bf);
  // out = y @ W_proj (fp32)
  gemm_bf16_kernel<0><<<dim3(CDIM / 128, (BATCH * TQ) / 128), 256, 0, stream>>>(
      y_bf, wp_bf, (float*)d_out, nullptr, nullptr, BATCH * TQ, CDIM, CDIM);

  (void)in_sizes; (void)n_in; (void)out_size; (void)ws_size;
}
